// KMLoss_71476845740241
// MI455X (gfx1250) — matrix operand order repaired and checked by reading, not compiled, not run on hardware
//
#include <hip/hip_runtime.h>

typedef __attribute__((ext_vector_type(16))) _Float16 v16h;
typedef __attribute__((ext_vector_type(8)))  float    v8f;

#define NQ   4096
#define NS   8192
#define DD   256
#define NC   64
#define NEG_INF_SENTINEL (-3.0e38f)
#define INF_ 1000.0f

// ---------------- workspace layout (bytes) ----------------
// Combined hi/lo fragment buffers: per fragment f, lanes' hi v16h at
// [f*64 + lane], lo v16h at [f*64 + 32 + lane]  (frag stride 2048 B).
// xq_frag : NQ*DD*4 = 4 MB  @ 0
// xs_frag : NS*DD*4 = 8 MB  @ 4 MB
// nq      : NQ*4    = 16 KB @ 12 MB
// ns      : NS*4    = 32 KB
// counts  : NC*4
// per_row : NQ*4
#define OFF_XQ     0
#define OFF_XS     (OFF_XQ + (size_t)NQ*DD*4)
#define OFF_NQ     (OFF_XS + (size_t)NS*DD*4)
#define OFF_NS     (OFF_NQ + (size_t)NQ*4)
#define OFF_CNT    (OFF_NS + (size_t)NS*4)
#define OFF_PERROW (OFF_CNT + (size_t)NC*4)

// ---------------- prep kernels ----------------

__global__ void zero_counts_kernel(int* counts) {
    if (threadIdx.x < NC) counts[threadIdx.x] = 0;
}

__global__ void count_kernel(const int* __restrict__ ys, int* __restrict__ counts) {
    int j = blockIdx.x * blockDim.x + threadIdx.x;
    if (j < NS) atomicAdd(&counts[ys[j]], 1);
}

__global__ void norm_kernel(const float* __restrict__ x, float* __restrict__ out, int rows) {
    int i = blockIdx.x * blockDim.x + threadIdx.x;
    if (i >= rows) return;
    const float4* p = (const float4*)(x + (size_t)i * DD);
    float s = 0.f;
#pragma unroll 8
    for (int q = 0; q < DD / 4; ++q) {
        float4 v = p[q];
        s += v.x * v.x + v.y * v.y + v.z * v.z + v.w * v.w;
    }
    out[i] = s;
}

// Swizzle xq rows into CDNA5 WMMA A-fragment layout (16-bit A, 16x32 MxK):
// lane l (l<16): row M=l, element e -> K = (e>>3)*16 + (l>>4)*8 + (e&7).
// hi fragment at [frag*64+lane], lo residual fragment at [frag*64+32+lane].
__global__ void swz_q_kernel(const float* __restrict__ xq, v16h* __restrict__ buf) {
    int t = blockIdx.x * blockDim.x + threadIdx.x;   // 0 .. NQ*DD/16-1 = 65535
    int lane  = t & 31;
    int frag  = t >> 5;            // mtile*8 + kstep
    int kstep = frag & 7;
    int mtile = frag >> 3;
    int m  = mtile * 16 + (lane & 15);
    int lh = lane >> 4;
    v16h h, l;
#pragma unroll
    for (int e = 0; e < 16; ++e) {
        int k = kstep * 32 + (e >> 3) * 16 + lh * 8 + (e & 7);
        float x = xq[(size_t)m * DD + k];
        _Float16 xh = (_Float16)x;
        h[e] = xh;
        l[e] = (_Float16)(x - (float)xh);
    }
    buf[(size_t)frag * 64 + lane]      = h;
    buf[(size_t)frag * 64 + 32 + lane] = l;
}

// Swizzle xs rows into CDNA5 WMMA B-fragment layout (16-bit B, 32x16 KxN):
// lane l: column N = l&15 ; element e -> K = (e>>3)*16 + (l>>4)*8 + (e&7).
// B[k][n] = xs[n][k] (xs.T).
__global__ void swz_s_kernel(const float* __restrict__ xs, v16h* __restrict__ buf) {
    int t = blockIdx.x * blockDim.x + threadIdx.x;   // 0 .. NS*DD/16-1 = 131071
    int lane  = t & 31;
    int frag  = t >> 5;            // ntile*8 + kstep
    int kstep = frag & 7;
    int ntile = frag >> 3;
    int n  = ntile * 16 + (lane & 15);
    int lh = lane >> 4;
    v16h h, l;
#pragma unroll
    for (int e = 0; e < 16; ++e) {
        int k = kstep * 32 + (e >> 3) * 16 + lh * 8 + (e & 7);
        float x = xs[(size_t)n * DD + k];
        _Float16 xh = (_Float16)x;
        h[e] = xh;
        l[e] = (_Float16)(x - (float)xh);
    }
    buf[(size_t)frag * 64 + lane]      = h;
    buf[(size_t)frag * 64 + 32 + lane] = l;
}

// ---------------- main kernel ----------------
// One block = 16 query rows, 8 wave32s. Each wave owns TWO 16-column
// N-subtiles per iteration (two independent WMMA accumulator chains for
// ILP); block covers 256 columns / iter, 32 iters. A fragments (8 k-steps,
// hi+lo = 128 VGPRs) stay resident across the whole N loop.
__global__ __launch_bounds__(256)
void main_kernel(const v16h* __restrict__ xq_frag, const v16h* __restrict__ xs_frag,
                 const float* __restrict__ nq, const float* __restrict__ ns,
                 const int* __restrict__ counts,
                 const int* __restrict__ yq, const int* __restrict__ ys,
                 const int* __restrict__ pos,
                 float* __restrict__ per_row) {
    __shared__ float s_sum[16][NC];    // per-row per-class logit sums
    __shared__ float s_max[16][128];   // per-row partial logsumexp (max)
    __shared__ float s_se[16][128];    // per-row partial logsumexp (sumexp)

    const int tid   = threadIdx.x;
    const int wave  = tid >> 5;
    const int lane  = tid & 31;
    const int laneN = lane & 15;
    const int laneH = lane >> 4;
    const int iBase = blockIdx.x * 16;

    for (int t = tid; t < 16 * NC; t += 256) ((float*)s_sum)[t] = 0.f;
    __syncthreads();

    // resident A fragments (combined hi/lo buffer, one base pointer)
    v16h a_hi[8], a_lo[8];
    {
        const v16h* baseA = xq_frag + (size_t)blockIdx.x * 8 * 64 + lane;
#pragma unroll
        for (int ks = 0; ks < 8; ++ks) {
            a_hi[ks] = baseA[ks * 64];
            a_lo[ks] = baseA[ks * 64 + 32];
        }
    }

    // per-row metadata for this lane's 8 rows (C-fragment rows: laneH*8 + r)
    float r_nq[8]; int r_yq[8]; int r_pos[8]; float r_diag[8];
#pragma unroll
    for (int r = 0; r < 8; ++r) {
        int i = iBase + laneH * 8 + r;
        r_nq[r]   = nq[i];
        r_yq[r]   = yq[i];
        r_pos[r]  = pos[i];
        r_diag[r] = (counts[r_yq[r]] > 1) ? -INF_ : 0.0f;
    }

    // per-lane online logsumexp state for the positive (same-class) term
    float mx[8], se[8];
#pragma unroll
    for (int r = 0; r < 8; ++r) { mx[r] = NEG_INF_SENTINEL; se[r] = 0.f; }

    for (int it = 0; it < NS / 256; ++it) {
        const int ntile0 = it * 16 + wave * 2;     // this wave's two subtiles
        const int j0     = ntile0 * 16 + laneN;
        const int j1     = j0 + 16;
        const float nsj0 = ns[j0];
        const float nsj1 = ns[j1];
        const int   ysj0 = ys[j0];
        const int   ysj1 = ys[j1];

        const v16h* baseB = xs_frag + (size_t)ntile0 * 8 * 64 + lane;

        v8f acc0 = {};
        v8f acc1 = {};
#pragma unroll
        for (int ks = 0; ks < 8; ++ks) {
            // subtile 0 fragments (immediate offsets off one base)
            v16h b0h = baseB[ks * 64];
            v16h b0l = baseB[ks * 64 + 32];
            // subtile 1 fragments (+1 ntile = +512 v16h)
            v16h b1h = baseB[512 + ks * 64];
            v16h b1l = baseB[512 + ks * 64 + 32];
            // split-precision: hi*hi + hi*lo + lo*hi  (~fp32 accuracy)
            acc0 = __builtin_amdgcn_wmma_f32_16x16x32_f16(false, a_hi[ks], false, b0h,
                                                          (short)0, acc0, false, false);
            acc1 = __builtin_amdgcn_wmma_f32_16x16x32_f16(false, a_hi[ks], false, b1h,
                                                          (short)0, acc1, false, false);
            acc0 = __builtin_amdgcn_wmma_f32_16x16x32_f16(false, a_hi[ks], false, b0l,
                                                          (short)0, acc0, false, false);
            acc1 = __builtin_amdgcn_wmma_f32_16x16x32_f16(false, a_hi[ks], false, b1l,
                                                          (short)0, acc1, false, false);
            acc0 = __builtin_amdgcn_wmma_f32_16x16x32_f16(false, a_lo[ks], false, b0h,
                                                          (short)0, acc0, false, false);
            acc1 = __builtin_amdgcn_wmma_f32_16x16x32_f16(false, a_lo[ks], false, b1h,
                                                          (short)0, acc1, false, false);
        }

#pragma unroll
        for (int r = 0; r < 8; ++r) {
            // subtile 0 element
            float sq0 = r_nq[r] + nsj0 - 2.0f * acc0[r];
            sq0 = fmaxf(sq0, 0.0f);
            float lg0 = -0.5f * sq0;                  // T = 1
            if (j0 == r_pos[r]) lg0 = r_diag[r];      // diagonal override
            atomicAdd(&s_sum[laneH * 8 + r][ysj0], lg0);
            if (ysj0 == r_yq[r]) {
                if (lg0 > mx[r]) { se[r] = se[r] * __expf(mx[r] - lg0) + 1.0f; mx[r] = lg0; }
                else             { se[r] += __expf(lg0 - mx[r]); }
            }
            // subtile 1 element
            float sq1 = r_nq[r] + nsj1 - 2.0f * acc1[r];
            sq1 = fmaxf(sq1, 0.0f);
            float lg1 = -0.5f * sq1;
            if (j1 == r_pos[r]) lg1 = r_diag[r];
            atomicAdd(&s_sum[laneH * 8 + r][ysj1], lg1);
            if (ysj1 == r_yq[r]) {
                if (lg1 > mx[r]) { se[r] = se[r] * __expf(mx[r] - lg1) + 1.0f; mx[r] = lg1; }
                else             { se[r] += __expf(lg1 - mx[r]); }
            }
        }
    }

    // dump per-lane (max, sumexp) partials
#pragma unroll
    for (int r = 0; r < 8; ++r) {
        int row = laneH * 8 + r;
        s_max[row][wave * 16 + laneN] = mx[r];
        s_se[row][wave * 16 + laneN]  = se[r];
    }
    __syncthreads();

    if (tid < 16) {
        const int row = tid;
        const int i   = iBase + row;
        // merge 128 logsumexp partials
        float M = NEG_INF_SENTINEL, S = 0.f;
        for (int t = 0; t < 128; ++t) {
            float m2 = s_max[row][t], s2 = s_se[row][t];
            if (m2 > M) { S = S * __expf(M - m2) + s2; M = m2; }
            else        { S += s2 * __expf(m2 - M); }
        }
        float pos_logit = M + __logf(S);

        // negative term: normalized per-class means, logsumexp over 64 classes
        int yqi = yq[i];
        float M2 = NEG_INF_SENTINEL;
        for (int c = 0; c < NC; ++c) {
            float adj = (float)counts[c] - ((yqi == c) ? 1.0f : 0.0f);
            float v = s_sum[row][c] / adj;
            M2 = fmaxf(M2, v);
        }
        float S2 = 0.f;
        for (int c = 0; c < NC; ++c) {
            float adj = (float)counts[c] - ((yqi == c) ? 1.0f : 0.0f);
            float v = s_sum[row][c] / adj;
            S2 += __expf(v - M2);
        }
        float neg_logit = M2 + __logf(S2);
        per_row[i] = neg_logit - pos_logit;
    }
}

__global__ void reduce_kernel(const float* __restrict__ per_row, float* __restrict__ out) {
    __shared__ float sh[256];
    float s = 0.f;
    for (int i = threadIdx.x; i < NQ; i += 256) s += per_row[i];
    sh[threadIdx.x] = s;
    __syncthreads();
    for (int st = 128; st > 0; st >>= 1) {
        if (threadIdx.x < st) sh[threadIdx.x] += sh[threadIdx.x + st];
        __syncthreads();
    }
    if (threadIdx.x == 0) out[0] = sh[0] / (float)NQ;
}

// ---------------- launch ----------------
extern "C" void kernel_launch(void* const* d_in, const int* in_sizes, int n_in,
                              void* d_out, int out_size, void* d_ws, size_t ws_size,
                              hipStream_t stream) {
    const float* xq  = (const float*)d_in[0];
    const int*   yq  = (const int*)  d_in[1];
    const float* xs  = (const float*)d_in[2];
    const int*   ys  = (const int*)  d_in[3];
    const int*   pos = (const int*)  d_in[4];
    float* out = (float*)d_out;

    char* w = (char*)d_ws;
    v16h*  xq_frag = (v16h*)(w + OFF_XQ);
    v16h*  xs_frag = (v16h*)(w + OFF_XS);
    float* nqv     = (float*)(w + OFF_NQ);
    float* nsv     = (float*)(w + OFF_NS);
    int*   counts  = (int*)  (w + OFF_CNT);
    float* per_row = (float*)(w + OFF_PERROW);

    zero_counts_kernel<<<1, 64, 0, stream>>>(counts);
    count_kernel<<<NS / 256, 256, 0, stream>>>(ys, counts);
    norm_kernel<<<(NQ + 255) / 256, 256, 0, stream>>>(xq, nqv, NQ);
    norm_kernel<<<(NS + 255) / 256, 256, 0, stream>>>(xs, nsv, NS);
    swz_q_kernel<<<(NQ * DD / 16) / 256, 256, 0, stream>>>(xq, xq_frag);
    swz_s_kernel<<<(NS * DD / 16) / 256, 256, 0, stream>>>(xs, xs_frag);
    main_kernel<<<NQ / 16, 256, 0, stream>>>(xq_frag, xs_frag,
                                             nqv, nsv, counts, yq, ys, pos, per_row);
    reduce_kernel<<<1, 256, 0, stream>>>(per_row, out);
}
